// SDCLayerSharedWeights_22892175688489
// MI455X (gfx1250) — matrix operand order repaired and checked by reading, not compiled, not run on hardware
//
#include <hip/hip_runtime.h>

#define Bn 4
#define Cn 128
#define Kn 128
#define Hn 256
#define Wn 256

typedef __attribute__((ext_vector_type(16))) __bf16 v16bf;
typedef __attribute__((ext_vector_type(8)))  float  v8f;

__device__ __forceinline__ unsigned short f2bf(float f) {
  unsigned u = __float_as_uint(f);
  u += 0x7FFFu + ((u >> 16) & 1u);          // round-to-nearest-even
  return (unsigned short)(u >> 16);
}

// ---------------------------------------------------------------------------
// Kernel 0: pre-swizzle fp32 weights into bf16 A-fragments laid out exactly in
// per-lane WMMA order. Fragment f = ((kh*3+kw)*4+cc)*8 + mtile is a 1KB block;
// lane L reads bytes [L*32, L*32+32). Within a lane, element j maps to
// K = j + (j>=8 ? 8 : 0) + (L>=16 ? 8 : 0)  (ISA 16-bit A 16x32 layout).
// ---------------------------------------------------------------------------
__global__ void sdc_swz_weights(const float* __restrict__ w,
                                unsigned short* __restrict__ ws) {
  int idx = blockIdx.x * 256 + threadIdx.x;        // 288 fragments * 512 elems
  if (idx >= 288 * 512) return;
  int j  = idx & 15;
  int L  = (idx >> 4) & 31;
  int f  = idx >> 9;
  int mt = f & 7;
  int cc = (f >> 3) & 3;
  int kw = (f >> 5) % 3;
  int kh = f / 96;
  int M    = L & 15;
  int half = L >> 4;
  int Kk = j + (j >= 8 ? 8 : 0) + half * 8;
  int k = mt * 16 + M;
  int c = cc * 32 + Kk;
  float val = w[((((size_t)k) * Cn + c) * 3 + kh) * 3 + kw];
  ws[idx] = f2bf(val);
}

// ---------------------------------------------------------------------------
// Main kernel: one workgroup per (b, h) output row. Implicit GEMM
// M=128 (out ch) x N=256 (cols) x K=1152 (C*9), bf16 WMMA, fp32 accum.
// Software-pipelined: next chunk's global loads are issued into a register
// staging buffer right after the current chunk drains to LDS, so HBM latency
// overlaps the blend VALU (Phase B) and WMMA (Phase C) work.
// ---------------------------------------------------------------------------
__global__ __launch_bounds__(256, 1)
void sdc_main_kernel(const float* __restrict__ x,
                     const unsigned short* __restrict__ wswz,
                     const float* __restrict__ bias,
                     float* __restrict__ out) {
  __shared__ float rawx[32][268];                  // 32 ch x (256 + 2*4 halo), padded
  __shared__ unsigned short xb[3][256][32];        // [kw][col][c32] bf16 B-tiles

  const int t     = threadIdx.x;
  const int b     = blockIdx.x >> 8;
  const int h     = blockIdx.x & 255;
  const int lane  = t & 31;
  const int wv    = t >> 5;                        // wave id == M-tile
  const int nmod  = lane & 15;
  const int nhalf = lane >> 4;

  // --- per-row blend weights (replicates numpy argsort/clamp logic) ---
  float rwv0 = 0.f, rwv1 = 0.f, rwv2 = 0.f, rwv3 = 0.f;
  {
    const float PI = 3.14159265358979323846f;
    float phi = -0.5f * PI + PI * (float)h / (float)(Hn - 1);
    float v = 1.0f / fabsf(cosf(phi));
    if (v >= 4.0f) {
      rwv3 = 1.0f;
    } else {
      int lo = (int)floorf(v);
      lo = lo < 1 ? 1 : (lo > 3 ? 3 : lo);
      float dlo = v - (float)lo;
      float dhi = (float)(lo + 1) - v;
      float invlo = (dlo == 0.0f) ? 10.0f : 1.0f / dlo;   // nearest==lo when tied
      float invhi = 1.0f / dhi;
      float s = invlo + invhi;
      float tmp[4] = {0.f, 0.f, 0.f, 0.f};
      tmp[lo - 1] = invlo / s;
      tmp[lo]     = invhi / s;
      rwv0 = tmp[0]; rwv1 = tmp[1]; rwv2 = tmp[2]; rwv3 = tmp[3];
    }
  }

  // valid vertical taps (clip image boundary up front)
  const int kh0   = (h == 0) ? 1 : 0;
  const int kh1   = (h == Hn - 1) ? 1 : 2;
  const int niter = (kh1 - kh0 + 1) * 4;           // (kh span) * 4 C-chunks

  auto issue_loads = [&](int r, int cc, float4* stage) {
    const float* xrow = x + ((((size_t)b) * Cn + cc * 32) * Hn + r) * Wn;
    #pragma unroll
    for (int i = 0; i < 8; ++i) {
      int idx = t + i * 256;
      int cp  = idx >> 6;
      int w4  = idx & 63;
      stage[i] = *(const float4*)(xrow + (size_t)cp * (Hn * Wn) + w4 * 4);
    }
  };

  v8f acc[16] = {};                                // 16 N-tiles, fp32 accum

  float4 stage[8];                                 // register staging (prefetch)
  issue_loads(h - 1 + kh0, 0, stage);              // prologue: chunk 0 in flight

  for (int it = 0; it < niter; ++it) {
    const int kh = kh0 + (it >> 2);
    const int cc = it & 3;

    __syncthreads();                               // prev Phase C done; LDS free
    // --- Phase A: drain staged registers into LDS, zero halos ---
    #pragma unroll
    for (int i = 0; i < 8; ++i) {
      int idx = t + i * 256;
      int cp  = idx >> 6;
      int w4  = idx & 63;
      *(float4*)&rawx[cp][4 + w4 * 4] = stage[i];
    }
    if (t < 32) {
      rawx[t][0] = 0.f; rawx[t][1] = 0.f; rawx[t][2] = 0.f; rawx[t][3] = 0.f;
      rawx[t][260] = 0.f; rawx[t][261] = 0.f; rawx[t][262] = 0.f; rawx[t][263] = 0.f;
    }
    // --- issue next chunk's global loads; they overlap Phases B and C ---
    if (it + 1 < niter) {
      int it2 = it + 1;
      int kh2 = kh0 + (it2 >> 2);
      int cc2 = it2 & 3;
      issue_loads(h - 1 + kh2, cc2, stage);
    }
    __syncthreads();

    // --- Phase B: build blended bf16 B-tiles for kw = -1, 0, +1 ---
    {
      int c2  = t & 15;                            // channel pair
      int wb  = t >> 4;                            // 16-column group
      int cp0 = c2 * 2;
      float r0[24], r1[24];
      #pragma unroll
      for (int i = 0; i < 6; ++i) {
        *(float4*)&r0[i * 4] = *(const float4*)&rawx[cp0][wb * 16 + i * 4];
        *(float4*)&r1[i * 4] = *(const float4*)&rawx[cp0 + 1][wb * 16 + i * 4];
      }
      #pragma unroll
      for (int k = 0; k < 16; ++k) {
        int w  = wb * 16 + k;
        int ci = k + 4;
        float c0 = r0[ci], c1 = r1[ci];
        float m0 = rwv0 * r0[ci - 1] + rwv1 * r0[ci - 2] + rwv2 * r0[ci - 3] + rwv3 * r0[ci - 4];
        float m1 = rwv0 * r1[ci - 1] + rwv1 * r1[ci - 2] + rwv2 * r1[ci - 3] + rwv3 * r1[ci - 4];
        float p0 = rwv0 * r0[ci + 1] + rwv1 * r0[ci + 2] + rwv2 * r0[ci + 3] + rwv3 * r0[ci + 4];
        float p1 = rwv0 * r1[ci + 1] + rwv1 * r1[ci + 2] + rwv2 * r1[ci + 3] + rwv3 * r1[ci + 4];
        *(unsigned int*)&xb[0][w][cp0] = (unsigned)f2bf(m0) | ((unsigned)f2bf(m1) << 16);
        *(unsigned int*)&xb[1][w][cp0] = (unsigned)f2bf(c0) | ((unsigned)f2bf(c1) << 16);
        *(unsigned int*)&xb[2][w][cp0] = (unsigned)f2bf(p0) | ((unsigned)f2bf(p1) << 16);
      }
    }
    __syncthreads();

    // --- Phase C: WMMA. Each wave: 1 A-frag + 16 B-frags + 16 WMMA per kw ---
    #pragma unroll
    for (int kw = 0; kw < 3; ++kw) {
      int f = ((kh * 3 + kw) * 4 + cc) * 8 + wv;
      union { uint4 u[2]; v16bf v; } A;
      const uint4* ap = (const uint4*)(wswz + (size_t)f * 512 + lane * 16);
      A.u[0] = ap[0];
      A.u[1] = ap[1];
      #pragma unroll
      for (int nt = 0; nt < 16; ++nt) {
        union { uint4 u[2]; v16bf v; } Bf;
        const uint4* bp = (const uint4*)&xb[kw][nt * 16 + nmod][nhalf * 8];
        Bf.u[0] = bp[0];
        Bf.u[1] = bp[2];
        acc[nt] = __builtin_amdgcn_wmma_f32_16x16x32_bf16(
            /*neg_a=*/false, A.v, /*neg_b=*/false, Bf.v,
            /*c_mod=*/(short)0, acc[nt], /*reuse_a=*/false, /*reuse_b=*/false);
      }
    }
  }

  // --- Epilogue: C/D layout = lane%16 -> N, VGPR v -> M = v + 8*(lane/16) ---
  float br[8];
  #pragma unroll
  for (int v = 0; v < 8; ++v)
    br[v] = bias[wv * 16 + v + 8 * nhalf];

  #pragma unroll
  for (int nt = 0; nt < 16; ++nt) {
    int wcol = nt * 16 + nmod;
    #pragma unroll
    for (int v = 0; v < 8; ++v) {
      int k = wv * 16 + v + 8 * nhalf;
      out[((((size_t)b) * Kn + k) * Hn + h) * Wn + wcol] = acc[nt][v] + br[v];
    }
  }
}

extern "C" void kernel_launch(void* const* d_in, const int* in_sizes, int n_in,
                              void* d_out, int out_size, void* d_ws, size_t ws_size,
                              hipStream_t stream) {
  (void)in_sizes; (void)n_in; (void)out_size; (void)ws_size;
  const float* x    = (const float*)d_in[0];
  const float* wgt  = (const float*)d_in[1];
  const float* bias = (const float*)d_in[2];
  float* out = (float*)d_out;
  unsigned short* wswz = (unsigned short*)d_ws;   // 288 KiB of bf16 fragments

  // 288 fragments * 512 elems = 147456 threads
  sdc_swz_weights<<<576, 256, 0, stream>>>(wgt, wswz);
  // one workgroup per (b, h)
  sdc_main_kernel<<<Bn * Hn, 256, 0, stream>>>(x, wswz, bias, out);
}
